// CausalSelfAttention_28750511079533
// MI455X (gfx1250) — matrix operand order repaired and checked by reading, not compiled, not run on hardware
//
#include <hip/hip_runtime.h>

// ---------------- problem constants ----------------
#define BB 2
#define TT 4096
#define CC 1024
#define HH 16
#define DD 64
#define C3 (3 * CC)

typedef __bf16 bf16;
typedef __attribute__((ext_vector_type(16))) __bf16 v16bf;
typedef __attribute__((ext_vector_type(8)))  __bf16 v8bf;
typedef __attribute__((ext_vector_type(4)))  __bf16 v4bf;
typedef __attribute__((ext_vector_type(8)))  float  v8f;
typedef __attribute__((ext_vector_type(4)))  int    v4i;
typedef __attribute__((ext_vector_type(8)))  int    v8i;

static __device__ __forceinline__ bf16 f2bf(float f) {
  unsigned u = __builtin_bit_cast(unsigned, f);
  u += 0x7FFFu + ((u >> 16) & 1u);          // round-to-nearest-even
  unsigned short h = (unsigned short)(u >> 16);
  return __builtin_bit_cast(bf16, h);
}

// LDS byte address of a generic pointer to __shared__ (flat low 32 bits = LDS addr)
static __device__ __forceinline__ unsigned lds_addr(const void* p) {
  return (unsigned)(size_t)p;
}

// async 16B global -> LDS copy (ASYNCcnt), one per lane
static __device__ __forceinline__ void async_copy_b128(unsigned lds, const void* gptr) {
  asm volatile("global_load_async_to_lds_b128 %0, %1, off"
               :: "v"(lds), "v"((unsigned long long)(size_t)gptr) : "memory");
}

// ---------------- f32 -> bf16 convert ----------------
__global__ __launch_bounds__(256) void cvt_f32_bf16(const float* __restrict__ src,
                                                    bf16* __restrict__ dst, int n) {
  int i = (blockIdx.x * 256 + threadIdx.x) * 4;   // n is a multiple of 1024
  float4 f = *(const float4*)(src + i);
  v4bf o;
  o[0] = f2bf(f.x); o[1] = f2bf(f.y); o[2] = f2bf(f.z); o[3] = f2bf(f.w);
  *(v4bf*)(dst + i) = o;
}

// ---------------- WMMA GEMM: C[M,N] = A[M,K] * Bw[N,K]^T ----------------
// Block tile 64(M) x 128(N), K-step 32. 256 threads = 8 waves in 2x4 grid,
// each wave computes a 32x32 tile = 2x2 v_wmma_f32_16x16x32_bf16.
// Double-buffered pipeline:
//   A tile   : GLOBAL_LOAD_ASYNC_TO_LDS_B128  (ASYNCcnt, in-order per wave)
//   B tile   : Tensor Data Mover transfer     (TENSORcnt, in-order per wave),
//              LDS-padded so rows land with stride LDB = 40 bf16 (16+4 dwords).
// Tile t+1 is issued before waiting for tile t; s_wait_{async,tensor}cnt 1
// retires exactly the older transfer while the newer stays in flight.
template <int WRITE_BF16>
__global__ __launch_bounds__(256) void gemm_bf16_wmma(const bf16* __restrict__ A,
                                                      const bf16* __restrict__ Bw,
                                                      void* __restrict__ Out,
                                                      int M, int N, int K) {
  constexpr int BM = 64, BN = 128, BK = 32, LDA = 40, LDB = 40;
  __shared__ __align__(16) bf16 As[2][BM * LDA];
  __shared__ __align__(16) bf16 Bs[2][BN * LDB];

  const int tid   = threadIdx.x;
  const int lane  = tid & 31, wid = tid >> 5;
  const int waveM = wid >> 2, waveN = wid & 3;
  const int bm = blockIdx.y * BM, bn = blockIdx.x * BN;
  const int l15 = lane & 15, g = lane >> 4;
  const int arow = tid >> 2, acol = (tid & 3) * 8;

  // ---- TDM descriptor (wave-uniform, SGPRs). Group1 is k-invariant. ----
  v8i g1;
  g1[0] = 0x00010000            // data_size = 1 (2 bytes)
        | (1 << 20)             // pad_enable
        | (3 << 22)             // pad_interval: 16 dwords (one 32-elem bf16 row)
        | (3 << 25);            // pad_amount: 4 dwords (16 B) -> LDS stride 80 B
  g1[1] = (K & 0xffff) << 16;                 // tensor_dim0[15:0]
  g1[2] = ((unsigned)K >> 16) | ((N & 0xffff) << 16);   // dim0 hi | tensor_dim1 lo
  g1[3] = (((unsigned)N >> 16) & 0xffff) | (BK << 16);  // dim1 hi | tile_dim0 = 32
  g1[4] = BN;                                  // tile_dim1 = 128 rows (tile_dim2 = 0)
  g1[5] = K;                                   // tensor_dim0_stride lo32
  g1[6] = 0;                                   // stride hi16 | dim1_stride lo16
  g1[7] = 0;

  // issue all transfers for K-tile t into LDS buffer `buf`
  auto issue_tile = [&](int t, int buf) {
    const int ks = t * BK;
    async_copy_b128(lds_addr(&As[buf][arow * LDA + acol]),
                    A + (size_t)(bm + arow) * K + ks + acol);
    if (wid == 0) {
      const unsigned long long gaddr =
          (unsigned long long)(size_t)Bw + ((size_t)bn * K + ks) * 2;
      v4i g0;
      g0[0] = 1;                                   // count = 1 (valid user D#)
      g0[1] = (int)lds_addr(&Bs[buf][0]);          // lds_addr
      g0[2] = (int)(gaddr & 0xffffffffu);          // global_addr[31:0]
      g0[3] = (int)(((gaddr >> 32) & 0x01ffffffu)  // global_addr[56:32]
                    | 0x80000000u);                // type = 2 ("image")
      asm volatile("tensor_load_to_lds %0, %1" :: "s"(g0), "s"(g1) : "memory");
    }
  };

  v8f acc[2][2];
  acc[0][0] = {}; acc[0][1] = {}; acc[1][0] = {}; acc[1][1] = {};

  const int NT = K / BK;
  issue_tile(0, 0);

  for (int t = 0; t < NT; ++t) {
    const int cur = t & 1;
    const bool more = (t + 1) < NT;
    if (more) issue_tile(t + 1, cur ^ 1);          // overlap next fetch w/ compute

    if (more) asm volatile("s_wait_asynccnt 1" ::: "memory");
    else      asm volatile("s_wait_asynccnt 0" ::: "memory");
    if (wid == 0) {
      if (more) __builtin_amdgcn_s_wait_tensorcnt(1);
      else      __builtin_amdgcn_s_wait_tensorcnt(0);
    }
    __syncthreads();                               // tile t visible to all waves

    // ---- A fragments (16x32, two 8-wide K runs per lane) ----
    v16bf af[2], bfr[2];
#pragma unroll
    for (int am = 0; am < 2; ++am) {
      const int r = waveM * 32 + am * 16 + l15;
      v8bf lo = *(const v8bf*)&As[cur][r * LDA + 8 * g];
      v8bf hi = *(const v8bf*)&As[cur][r * LDA + 8 * g + 16];
#pragma unroll
      for (int e = 0; e < 8; ++e) { af[am][e] = lo[e]; af[am][e + 8] = hi[e]; }
    }
    // ---- B fragments (col per lane, two 8-wide K runs per lane) ----
#pragma unroll
    for (int bnn = 0; bnn < 2; ++bnn) {
      const int c = waveN * 32 + bnn * 16 + l15;
      v8bf lo = *(const v8bf*)&Bs[cur][c * LDB + 8 * g];
      v8bf hi = *(const v8bf*)&Bs[cur][c * LDB + 8 * g + 16];
#pragma unroll
      for (int e = 0; e < 8; ++e) { bfr[bnn][e] = lo[e]; bfr[bnn][e + 8] = hi[e]; }
    }
#pragma unroll
    for (int am = 0; am < 2; ++am)
#pragma unroll
      for (int bnn = 0; bnn < 2; ++bnn)
        acc[am][bnn] = __builtin_amdgcn_wmma_f32_16x16x32_bf16(
            false, af[am], false, bfr[bnn], (short)0, acc[am][bnn], false, false);

    __syncthreads();       // all waves done reading buf `cur` -> next iter may fill it
  }

  // ---- epilogue: C/D layout is M = r + 8*g, N = l15 ----
#pragma unroll
  for (int am = 0; am < 2; ++am)
#pragma unroll
    for (int bnn = 0; bnn < 2; ++bnn) {
      const int col = bn + waveN * 32 + bnn * 16 + l15;
#pragma unroll
      for (int r = 0; r < 8; ++r) {
        const int row = bm + waveM * 32 + am * 16 + r + 8 * g;
        if (WRITE_BF16)
          ((bf16*)Out)[(size_t)row * N + col] = f2bf(acc[am][bnn][r]);
        else
          ((float*)Out)[(size_t)row * N + col] = acc[am][bnn][r];
      }
    }
}

// ---------------- flash attention (causal), one wave = 16 query rows ----------------
// qkv: [B*T, 3C] bf16; Q at +h*D, K at +C+h*D, V at +2C+h*D. out: [B,T,C] bf16.
// V tile is copied global->LDS asynchronously (natural [key][d] layout) and
// overlapped with the S = QK^T WMMAs and the softmax VALU work.
__global__ __launch_bounds__(256) void flash_attn_wmma(const bf16* __restrict__ qkv,
                                                       bf16* __restrict__ out) {
  const int tid = threadIdx.x, lane = tid & 31, wid = tid >> 5;
  const int l15 = lane & 15, g = lane >> 4;
  const int h = blockIdx.y, b = blockIdx.z;
  const int q0 = blockIdx.x * 128 + wid * 16;

  // per-wave LDS: P tile (16x32 bf16 = 512) + V tile (32x64 bf16 = 2048)
  __shared__ __align__(16) bf16 smem[8 * 2560];
  bf16* Pl = smem + wid * 2560;
  bf16* Vl = Pl + 512;
  const unsigned vbase = lds_addr(Vl);

  const size_t rs = (size_t)C3;
  const bf16* Qp = qkv + (size_t)b * TT * rs + h * DD;
  const bf16* Kp = Qp + CC;
  const bf16* Vp = Qp + 2 * CC;

  // resident Q fragments: A-matrix 16x32 for d=[0,32) and [32,64)
  v16bf qf[2];
#pragma unroll
  for (int kk = 0; kk < 2; ++kk) {
    const bf16* p = Qp + (size_t)(q0 + l15) * rs + kk * 32 + 8 * g;
    v8bf lo = *(const v8bf*)p;
    v8bf hi = *(const v8bf*)(p + 16);
#pragma unroll
    for (int e = 0; e < 8; ++e) { qf[kk][e] = lo[e]; qf[kk][e + 8] = hi[e]; }
  }

  float m[8], l[8];
  v8f acc[4];
#pragma unroll
  for (int r = 0; r < 8; ++r) { m[r] = -3.0e30f; l[r] = 0.f; }
  acc[0] = {}; acc[1] = {}; acc[2] = {}; acc[3] = {};

  for (int kt = 0; kt < q0 + 16; kt += 32) {     // wave-uniform causal bound
    // ---- kick off async V tile copy: Vl[key][d], 16B per lane-iteration ----
#pragma unroll
    for (int it = 0; it < 8; ++it) {
      const int key = it * 4 + (lane >> 3);
      const int dc = (lane & 7) * 8;
      async_copy_b128(vbase + (unsigned)(key * DD + dc) * 2,
                      Vp + (size_t)(kt + key) * rs + dc);
    }

    // ---- S = Q K^T : two 16(q) x 16(key) tiles, K-dim = D = 64 ----
    v8f s[2];
#pragma unroll
    for (int j = 0; j < 2; ++j) {
      v8f z = {};
#pragma unroll
      for (int kk = 0; kk < 2; ++kk) {
        const bf16* p = Kp + (size_t)(kt + j * 16 + l15) * rs + kk * 32 + 8 * g;
        v8bf lo = *(const v8bf*)p;
        v8bf hi = *(const v8bf*)(p + 16);
        v16bf kf;
#pragma unroll
        for (int e = 0; e < 8; ++e) { kf[e] = lo[e]; kf[e + 8] = hi[e]; }
        z = __builtin_amdgcn_wmma_f32_16x16x32_bf16(false, qf[kk], false, kf,
                                                    (short)0, z, false, false);
      }
      s[j] = z;
    }

    // ---- online softmax (row M = r + 8*g lives across 16 lanes of a half) ----
    float p0[8], p1[8];
#pragma unroll
    for (int r = 0; r < 8; ++r) {
      const int qrow = q0 + r + 8 * g;
      float s0 = s[0][r] * 0.125f;               // 1/sqrt(64)
      float s1 = s[1][r] * 0.125f;
      if (kt + l15 > qrow)      s0 = -3.0e30f;   // causal mask
      if (kt + 16 + l15 > qrow) s1 = -3.0e30f;
      float mx = fmaxf(s0, s1);
      mx = fmaxf(mx, __shfl_xor(mx, 1, 16));
      mx = fmaxf(mx, __shfl_xor(mx, 2, 16));
      mx = fmaxf(mx, __shfl_xor(mx, 4, 16));
      mx = fmaxf(mx, __shfl_xor(mx, 8, 16));
      const float mn = fmaxf(m[r], mx);
      const float alpha = __expf(m[r] - mn);
      const float e0 = __expf(s0 - mn);
      const float e1 = __expf(s1 - mn);
      float rsum = e0 + e1;
      rsum += __shfl_xor(rsum, 1, 16);
      rsum += __shfl_xor(rsum, 2, 16);
      rsum += __shfl_xor(rsum, 4, 16);
      rsum += __shfl_xor(rsum, 8, 16);
      l[r] = l[r] * alpha + rsum;
      m[r] = mn;
#pragma unroll
      for (int d = 0; d < 4; ++d) acc[d][r] *= alpha;
      p0[r] = e0; p1[r] = e1;
    }

    // ---- C-layout P -> LDS -> A-layout fragment ----
#pragma unroll
    for (int r = 0; r < 8; ++r) {
      const int row = r + 8 * g;
      Pl[row * 32 + l15]      = f2bf(p0[r]);
      Pl[row * 32 + 16 + l15] = f2bf(p1[r]);
    }
    asm volatile("s_wait_dscnt 0" ::: "memory");   // per-wave LDS ordering

    v16bf pa;
    {
      const bf16* p = Pl + l15 * 32 + 8 * g;
      v8bf lo = *(const v8bf*)p;
      v8bf hi = *(const v8bf*)(p + 16);
#pragma unroll
      for (int e = 0; e < 8; ++e) { pa[e] = lo[e]; pa[e + 8] = hi[e]; }
    }

    // ---- V tile has landed by now (overlapped with the work above) ----
    asm volatile("s_wait_asynccnt 0" ::: "memory");

    // ---- O += P(16x32) * V(32x64): gather B-fragments from LDS, 4 WMMAs ----
#pragma unroll
    for (int d = 0; d < 4; ++d) {
      v16bf vf;
#pragma unroll
      for (int e = 0; e < 16; ++e)
        vf[e] = Vl[(8 * g + (e < 8 ? e : e + 8)) * DD + d * 16 + l15];
      acc[d] = __builtin_amdgcn_wmma_f32_16x16x32_bf16(false, pa, false, vf,
                                                       (short)0, acc[d], false, false);
    }
  }

  // ---- normalize and write attn output (bf16, [B,T,C]) ----
#pragma unroll
  for (int d = 0; d < 4; ++d)
#pragma unroll
    for (int r = 0; r < 8; ++r) {
      const int qrow = q0 + r + 8 * g;
      const float v = acc[d][r] / l[r];
      out[((size_t)b * TT + qrow) * CC + h * DD + d * 16 + l15] = f2bf(v);
    }
}

// ---------------- launcher ----------------
extern "C" void kernel_launch(void* const* d_in, const int* in_sizes, int n_in,
                              void* d_out, int out_size, void* d_ws, size_t ws_size,
                              hipStream_t stream) {
  const float* x    = (const float*)d_in[0];   // [B,T,C]
  const float* Wqkv = (const float*)d_in[1];   // [3C,C]
  const float* Wout = (const float*)d_in[2];   // [C,C]
  float* out = (float*)d_out;

  char* ws = (char*)d_ws;
  bf16* xb    = (bf16*)(ws + 0);           // 16,777,216 B
  bf16* wqkvb = (bf16*)(ws + 16777216);    //  6,291,456 B
  bf16* woutb = (bf16*)(ws + 23068672);    //  2,097,152 B
  bf16* qkvb  = (bf16*)(ws + 25165824);    // 50,331,648 B
  bf16* attnb = (bf16*)(ws + 75497472);    // 16,777,216 B   (total ~92.3 MB)

  const int M = BB * TT;                   // 8192
  cvt_f32_bf16<<<(M * CC) / 1024, 256, 0, stream>>>(x, xb, M * CC);
  cvt_f32_bf16<<<(C3 * CC) / 1024, 256, 0, stream>>>(Wqkv, wqkvb, C3 * CC);
  cvt_f32_bf16<<<(CC * CC) / 1024, 256, 0, stream>>>(Wout, woutb, CC * CC);

  dim3 g1(C3 / 128, M / 64);
  gemm_bf16_wmma<1><<<g1, 256, 0, stream>>>(xb, wqkvb, (void*)qkvb, M, C3, CC);

  dim3 g2(TT / 128, HH, BB);
  flash_attn_wmma<<<g2, 256, 0, stream>>>(qkvb, attnb);

  dim3 g3(CC / 128, M / 64);
  gemm_bf16_wmma<0><<<g3, 256, 0, stream>>>(attnb, woutb, (void*)out, M, CC, CC);
}
